// RNN_76321568850240
// MI455X (gfx1250) — matrix operand order repaired and checked by reading, not compiled, not run on hardware
//
#include <hip/hip_runtime.h>

// Problem constants (match the reference)
#define NB  256    // batch B
#define TT  100    // timesteps T
#define NG  4096   // hidden size Ng
#define NP  512    // place cells Np
#define DV  2      // input dim

typedef __bf16 bf16_t;
typedef __attribute__((ext_vector_type(16))) __bf16 v16bf;
typedef __attribute__((ext_vector_type(8)))  float  v8f;
typedef __attribute__((ext_vector_type(4)))  unsigned int u32x4;

union FragBF {
    v16bf v;
    u32x4 q[2];
};

// ---------------------------------------------------------------------------
// Generic NT bf16 WMMA GEMM:  C[M x N] = op( A[M x K] * Bt[N x K]^T )
//   A, Bt are bf16 row-major with K contiguous (leading dims lda/ldb).
//   Optional fused input projection  x += v[b,t,0]*Win[0,n] + v[b,t,1]*Win[1,n]
//   and relu.  OutT is __bf16 (hidden state) or float (final predictions).
//
// Fragment addressing follows the CDNA5 ISA VGPR layouts:
//   A 16x32 bf16: lane m=lane%16, half=lane/16; VGPR0..3 = K[half*8 .. +7],
//                 VGPR4..7 = K[16+half*8 .. +7]  -> two contiguous 16B loads.
//   B 32x16 bf16: lane n=lane%16, half=lane/16; VGPR0..3 = K[half*8 .. +7],
//                 VGPR4..7 = K[16+half*8 .. +7]  -> two contiguous 16B loads.
//   C/D f32:      element r -> row (r + 8*half), col lane%16.
// ---------------------------------------------------------------------------
template<int BM, int BN, int WM, int WN, bool RELU, bool FUSE_IN, typename OutT>
__global__ __launch_bounds__((BM/WM)*(BN/WN)*32)
void gemm_nt_bf16(const bf16_t* __restrict__ A, int lda,
                  const bf16_t* __restrict__ Bt, int ldb,
                  OutT* __restrict__ C, int ldc, int K,
                  const float* __restrict__ vin,   // [B,T,DV] (FUSE_IN only)
                  const float* __restrict__ Win,   // [DV,NG]  (FUSE_IN only)
                  int t)
{
    constexpr int MT = WM / 16;        // M fragments per wave
    constexpr int NTf = WN / 16;       // N fragments per wave
    constexpr int WAVES_M = BM / WM;

    const int lane = threadIdx.x & 31;
    const int wid  = threadIdx.x >> 5;
    const int wm   = wid % WAVES_M;
    const int wn   = wid / WAVES_M;
    const int m0   = blockIdx.x * BM + wm * WM;
    const int n0   = blockIdx.y * BN + wn * WN;
    const int half = lane >> 4;
    const int l16  = lane & 15;

    v8f acc[MT][NTf];
    #pragma unroll
    for (int i = 0; i < MT; ++i)
        #pragma unroll
        for (int j = 0; j < NTf; ++j)
            acc[i][j] = (v8f){};

    const bf16_t* pa0[MT];
    const bf16_t* pb0[NTf];
    #pragma unroll
    for (int i = 0; i < MT; ++i)
        pa0[i] = A + (size_t)(m0 + i * 16 + l16) * lda + half * 8;
    #pragma unroll
    for (int j = 0; j < NTf; ++j)
        pb0[j] = Bt + (size_t)(n0 + j * 16 + l16) * ldb + half * 8;

    for (int k = 0; k < K; k += 32) {
        FragBF a[MT], b[NTf];
        #pragma unroll
        for (int i = 0; i < MT; ++i) {
            a[i].q[0] = *(const u32x4*)(pa0[i] + k);        // K = half*8   .. +7
            a[i].q[1] = *(const u32x4*)(pa0[i] + k + 16);   // K = 16+half*8.. +7
        }
        #pragma unroll
        for (int j = 0; j < NTf; ++j) {
            b[j].q[0] = *(const u32x4*)(pb0[j] + k);        // K = half*8   .. +7
            b[j].q[1] = *(const u32x4*)(pb0[j] + k + 16);   // K = 16+half*8.. +7
        }
        #pragma unroll
        for (int i = 0; i < MT; ++i)
            #pragma unroll
            for (int j = 0; j < NTf; ++j)
                acc[i][j] = __builtin_amdgcn_wmma_f32_16x16x32_bf16(
                    false, a[i].v, false, b[j].v,
                    (short)0, acc[i][j], false, false);
    }

    // Epilogue: optional fused input projection + relu, then store.
    #pragma unroll
    for (int i = 0; i < MT; ++i) {
        float x0[8], x1[8];
        if (FUSE_IN) {
            #pragma unroll
            for (int r = 0; r < 8; ++r) {
                const int mm = m0 + i * 16 + half * 8 + r;   // batch row b
                x0[r] = vin[((size_t)mm * TT + t) * DV + 0];
                x1[r] = vin[((size_t)mm * TT + t) * DV + 1];
            }
        }
        #pragma unroll
        for (int j = 0; j < NTf; ++j) {
            const int nn = n0 + j * 16 + l16;
            float w0 = 0.f, w1 = 0.f;
            if (FUSE_IN) { w0 = Win[nn]; w1 = Win[NG + nn]; }
            #pragma unroll
            for (int r = 0; r < 8; ++r) {
                const int mm = m0 + i * 16 + half * 8 + r;
                float x = acc[i][j][r];
                if (FUSE_IN) x += x0[r] * w0 + x1[r] * w1;
                if (RELU)    x = fmaxf(x, 0.f);
                C[(size_t)mm * ldc + nn] = (OutT)x;
            }
        }
    }
}

// ---------------------------------------------------------------------------
// Tiled transpose + f32 -> bf16 convert:  out[C x R] = bf16(in[R x C]^T)
// ---------------------------------------------------------------------------
__global__ void transpose_cvt(const float* __restrict__ in, bf16_t* __restrict__ out,
                              int R, int C)
{
    __shared__ float tile[32][33];
    const int c0 = blockIdx.x * 32;
    const int r0 = blockIdx.y * 32;
    const int tx = threadIdx.x;
    const int ty = threadIdx.y;        // blockDim = (32, 8)
    #pragma unroll
    for (int k = 0; k < 32; k += 8)
        tile[ty + k][tx] = in[(size_t)(r0 + ty + k) * C + (c0 + tx)];
    __syncthreads();
    #pragma unroll
    for (int k = 0; k < 32; k += 8)
        out[(size_t)(c0 + ty + k) * R + (r0 + tx)] = (bf16_t)tile[tx][ty + k];
}

// Elementwise f32 -> bf16
__global__ void cvt_bf16(const float* __restrict__ in, bf16_t* __restrict__ out, int n)
{
    const int i = blockIdx.x * blockDim.x + threadIdx.x;
    if (i < n) out[i] = (bf16_t)in[i];
}

// ---------------------------------------------------------------------------
extern "C" void kernel_launch(void* const* d_in, const int* in_sizes, int n_in,
                              void* d_out, int out_size, void* d_ws, size_t ws_size,
                              hipStream_t stream)
{
    const float* v    = (const float*)d_in[0];  // [B,T,DV]
    const float* P0   = (const float*)d_in[1];  // [B,NP]
    const float* Wenc = (const float*)d_in[2];  // [NP,NG]
    const float* Win  = (const float*)d_in[3];  // [DV,NG]
    const float* Wrec = (const float*)d_in[4];  // [NG,NG]
    const float* Wdec = (const float*)d_in[5];  // [NG,NP]
    float* out = (float*)d_out;                 // [B,T,NP]

    // Workspace carve-up (~46.5 MB total)
    char* ws = (char*)d_ws;
    bf16_t* WrT = (bf16_t*)ws;  ws += (size_t)NG * NG * sizeof(bf16_t); // [g][k]
    bf16_t* WdT = (bf16_t*)ws;  ws += (size_t)NP * NG * sizeof(bf16_t); // [p][g]
    bf16_t* WeT = (bf16_t*)ws;  ws += (size_t)NG * NP * sizeof(bf16_t); // [g][p]
    bf16_t* P0b = (bf16_t*)ws;  ws += (size_t)NB * NP * sizeof(bf16_t); // [b][p]
    bf16_t* h0  = (bf16_t*)ws;  ws += (size_t)NB * NG * sizeof(bf16_t); // ping
    bf16_t* h1  = (bf16_t*)ws;                                          // pong
    bf16_t* h[2] = { h0, h1 };

    // --- one-time weight prep: transpose to NT layout + bf16 convert --------
    dim3 tb(32, 8);
    transpose_cvt<<<dim3(NG/32, NG/32), tb, 0, stream>>>(Wrec, WrT, NG, NG);
    transpose_cvt<<<dim3(NP/32, NG/32), tb, 0, stream>>>(Wdec, WdT, NG, NP);
    transpose_cvt<<<dim3(NG/32, NP/32), tb, 0, stream>>>(Wenc, WeT, NP, NG);
    cvt_bf16<<<(NB*NP + 255)/256, 256, 0, stream>>>(P0, P0b, NB * NP);

    // --- encoder: h[0] = P0 @ W_enc  (M=256, N=4096, K=512) -----------------
    // 64x64 block tile, 4 waves -> 256 workgroups for occupancy.
    gemm_nt_bf16<64,64,32,32,false,false,bf16_t>
        <<<dim3(NB/64, NG/64), 128, 0, stream>>>(
            P0b, NP, WeT, NP, h[0], NG, NP, nullptr, nullptr, 0);

    // --- recurrence + per-step decode --------------------------------------
    for (int t = 0; t < TT; ++t) {
        bf16_t* hprev = h[t & 1];
        bf16_t* hnext = h[(t + 1) & 1];

        // h_next = relu(v[:,t,:] @ W_in + h_prev @ W_rec)   (M=256,N=4096,K=4096)
        // 64x64 block tile, 4 waves -> 256 WGs per step (latency-critical path).
        gemm_nt_bf16<64,64,32,32,true,true,bf16_t>
            <<<dim3(NB/64, NG/64), 128, 0, stream>>>(
                hprev, NG, WrT, NG, hnext, NG, NG, v, Win, t);

        // place_preds[:,t,:] = h_next @ W_dec               (M=256,N=512,K=4096)
        // 32x64 block tile, 2 waves -> 64 WGs.
        gemm_nt_bf16<32,64,32,32,false,false,float>
            <<<dim3(NB/32, NP/64), 64, 0, stream>>>(
                hnext, NG, WdT, NG, out + (size_t)t * NP, TT * NP, NG,
                nullptr, nullptr, 0);
    }
}
